// MoELayer_89824946029282
// MI455X (gfx1250) — compile-verified, matrix operand order checked by reading, not run on hardware
//
#include <hip/hip_runtime.h>
#include <math.h>

// ---------------- problem constants ----------------
#define DIMC 1024
#define NTOK 2048
#define EXP  8
#define FFD  2048          // FF
#define TWOFF 4096         // 2*FF
#define TILE_M 32          // tokens per workgroup tile (2 row-tiles of 16)
#define MB 512             // ff macro-block width (acts staged in LDS)
#define NTHREADS 256       // 8 waves (wave32)
#define XPAD 8             // LDS row padding (halfs) -> conflict-free b128
#define XSTR (DIMC + XPAD)
#define ASTR (MB + XPAD)

typedef __attribute__((ext_vector_type(16))) __bf16 v16bf;
typedef __attribute__((ext_vector_type(8)))  float  v8f;

// ---- A-fragment (16x32 bf16) from row-major bf16 LDS -------------------
// lane L: row M = L%16 ; K = [o..o+7] and [16+o..16+o+7], o = 8*(L/16).
// Two contiguous 16B runs -> ds_load_b128 x2 (rows padded: conflict-free).
__device__ __forceinline__ v16bf load_a_lds(const __bf16* p) {
  v16bf r;
#pragma unroll
  for (int i = 0; i < 8; ++i) { r[i] = p[i]; r[i + 8] = p[i + 16]; }
  return r;
}

// ---- B-fragment (32x16 bf16) gathered from row-major fp32 global -------
// lane L: col N = n0 + L%16 ; K pattern as A. For fixed element i, lanes
// 0-15 read 16 consecutive floats (64B coalesced), lanes 16-31 the K+8 row.
// f32 -> bf16 via packed VALU cvt. Each fragment feeds 2 WMMAs (M halves).
__device__ __forceinline__ v16bf load_b_gbl(const float* __restrict__ w, int ld,
                                            int k0, int n0, int nsub, int ko) {
  const float* p  = w + (size_t)(k0 + ko) * ld + n0 + nsub;
  const float* p2 = p + (size_t)16 * ld;
  v16bf r;
#pragma unroll
  for (int i = 0; i < 8; ++i) r[i] = (__bf16)p[(size_t)i * ld];
#pragma unroll
  for (int i = 0; i < 8; ++i) r[i + 8] = (__bf16)p2[(size_t)i * ld];
  return r;
}

// =======================================================================
// Kernel 1: gating. One block per token: logits, top-2 softmax weights,
// full router softmax (stored for deterministic LB reduce), bucket append.
// =======================================================================
__global__ __launch_bounds__(128) void gate_kernel(
    const float* __restrict__ x, const float* __restrict__ gw,
    const float* __restrict__ gb, int* __restrict__ cnt,
    int* __restrict__ bucket, float* __restrict__ cwbuf,
    float* __restrict__ probs) {
  __shared__ float sh[EXP * 128];
  __shared__ float logits[EXP];
  const int n = blockIdx.x, tid = threadIdx.x;
  const float* xr = x + (size_t)n * DIMC;
  float p[EXP];
#pragma unroll
  for (int e = 0; e < EXP; ++e) p[e] = 0.f;
  for (int c = tid; c < DIMC; c += 128) {
    float xv = xr[c];
#pragma unroll
    for (int e = 0; e < EXP; ++e) p[e] += xv * gw[e * DIMC + c];
  }
#pragma unroll
  for (int e = 0; e < EXP; ++e) sh[e * 128 + tid] = p[e];
  __syncthreads();
  if (tid < EXP) {
    float s = 0.f;
    for (int i = 0; i < 128; ++i) s += sh[tid * 128 + i];
    logits[tid] = s + gb[tid];
  }
  __syncthreads();
  if (tid == 0) {
    float l[EXP];
#pragma unroll
    for (int e = 0; e < EXP; ++e) l[e] = logits[e];
    int i1 = 0;
    for (int e = 1; e < EXP; ++e) if (l[e] > l[i1]) i1 = e;      // first max
    int i2 = (i1 == 0) ? 1 : 0;
    for (int e = 0; e < EXP; ++e) if (e != i1 && l[e] > l[i2]) i2 = e;
    float m  = l[i1];
    float e2 = expf(l[i2] - m);
    float wA = 1.f / (1.f + e2);          // softmax over the top-2 pair
    float wB = e2 / (1.f + e2);
    float se = 0.f, pe[EXP];
#pragma unroll
    for (int e = 0; e < EXP; ++e) { pe[e] = expf(l[e] - m); se += pe[e]; }
#pragma unroll
    for (int e = 0; e < EXP; ++e) probs[(size_t)n * EXP + e] = pe[e] / se;
    int p1 = atomicAdd(&cnt[i1], 1);
    bucket[i1 * NTOK + p1] = n;  cwbuf[i1 * NTOK + p1] = wA;
    int p2 = atomicAdd(&cnt[i2], 1);
    bucket[i2 * NTOK + p2] = n;  cwbuf[i2 * NTOK + p2] = wB;
  }
}

// =======================================================================
// Kernel 2: sparse expert FFN (bf16 WMMA, f32 accumulate).
// grid.x = EXP * 64 tiles; 256 threads = 8 waves; TILE_M = 32 tokens.
// Each wave: phase1 -> 4 ff-chunks/macro-block (both M halves),
//            phase2 -> 128 output columns (8 col-tiles x both M halves).
// =======================================================================
__global__ __launch_bounds__(NTHREADS) void moe_ffn_kernel(
    const float* __restrict__ x,
    const float* __restrict__ w1, const float* __restrict__ b1,
    const float* __restrict__ w2, const float* __restrict__ b2,
    const int* __restrict__ cnt, const int* __restrict__ bucket,
    const float* __restrict__ cwbuf, float* __restrict__ y) {
  __shared__ __bf16 xs[TILE_M][XSTR];    // ~64.5 KB : x tile in bf16 (padded)
  __shared__ __bf16 acts[TILE_M][ASTR];  // ~32.5 KB : GELU-gated (a*gelu(g)) macro-block
  __shared__ int   stok[TILE_M];
  __shared__ float scw[TILE_M];

  const int e    = blockIdx.x >> 6;      // 64 tiles per expert
  const int tile = blockIdx.x & 63;
  const int count = cnt[e];
  const int t0 = tile * TILE_M;
  if (t0 >= count) return;               // uniform across the block

  const int tid  = threadIdx.x;
  const int lane = tid & 31;
  const int wave = tid >> 5;
  const int nsub = lane & 15;            // row (A) or column (B/D) in tile
  const int ko   = (lane >> 4) * 8;      // K / M half offset

  if (tid < TILE_M) {
    int idx = t0 + tid;
    bool v = idx < count;
    stok[tid] = v ? bucket[e * NTOK + idx] : bucket[e * NTOK + t0];
    scw[tid]  = v ? cwbuf[e * NTOK + idx] : 0.f;   // padded rows contribute 0
  }
  __syncthreads();
  // stage x tile (gathered rows) as bf16, coalesced along C
  for (int i = tid; i < TILE_M * DIMC; i += NTHREADS) {
    int m = i >> 10, c = i & (DIMC - 1);
    xs[m][c] = (__bf16)x[(size_t)stok[m] * DIMC + c];
  }
  const float* w1e = w1 + (size_t)e * DIMC * TWOFF;
  const float* w2e = w2 + (size_t)e * FFD * DIMC;

  const v8f z8 = {0.f, 0.f, 0.f, 0.f, 0.f, 0.f, 0.f, 0.f};
  v8f oacc[16];                          // [8 col-tiles][2 M-halves] f32
#pragma unroll
  for (int i = 0; i < 16; ++i) oacc[i] = z8;
  __syncthreads();

  for (int mb = 0; mb < FFD / MB; ++mb) {
    // ---- phase 1: act[:, mb*MB .. +MB) = a * gelu(g) ------------------
    for (int jl = wave; jl < MB / 16; jl += 8) {     // 4 chunks per wave
      const int fcol = mb * MB + jl * 16;
      v8f a0 = z8, a1 = z8, g0 = z8, g1 = z8;
      for (int k0 = 0; k0 < DIMC; k0 += 32) {
        v16bf af0 = load_a_lds(&xs[nsub][k0 + ko]);        // rows 0..15
        v16bf af1 = load_a_lds(&xs[16 + nsub][k0 + ko]);   // rows 16..31
        v16bf bA  = load_b_gbl(w1e, TWOFF, k0, fcol,       nsub, ko);
        v16bf bG  = load_b_gbl(w1e, TWOFF, k0, FFD + fcol, nsub, ko);
        if (k0 + 32 < DIMC) {   // cover gather latency of next k-step
          __builtin_prefetch(w1e + (size_t)(k0 + 32 + ko) * TWOFF + fcol + nsub, 0, 0);
          __builtin_prefetch(w1e + (size_t)(k0 + 32 + ko) * TWOFF + FFD + fcol + nsub, 0, 0);
        }
        a0 = __builtin_amdgcn_wmma_f32_16x16x32_bf16(false, af0, false, bA,
                                                     (short)0, a0, false, false);
        a1 = __builtin_amdgcn_wmma_f32_16x16x32_bf16(false, af1, false, bA,
                                                     (short)0, a1, false, false);
        g0 = __builtin_amdgcn_wmma_f32_16x16x32_bf16(false, af0, false, bG,
                                                     (short)0, g0, false, false);
        g1 = __builtin_amdgcn_wmma_f32_16x16x32_bf16(false, af1, false, bG,
                                                     (short)0, g1, false, false);
      }
      const float ba = b1[e * TWOFF + fcol + nsub];
      const float bg = b1[e * TWOFF + FFD + fcol + nsub];
#pragma unroll
      for (int r = 0; r < 2; ++r) {
        const v8f av8 = r ? a1 : a0;
        const v8f gv8 = r ? g1 : g0;
#pragma unroll
        for (int i = 0; i < 8; ++i) {
          int m = i + ko + 16 * r;                 // D-layout: M = r8 + 8*half
          float av = av8[i] + ba;
          float gv = gv8[i] + bg;
          float ge = 0.5f * gv * (1.f + erff(gv * 0.70710678118654752f));
          acts[m][jl * 16 + nsub] = (__bf16)(av * ge);
        }
      }
    }
    __syncthreads();
    // ---- phase 2: oacc += acts(32xMB) @ w2[mb*MB.., wave's 128 cols] --
#pragma unroll
    for (int ct = 0; ct < 8; ++ct) {
      const int c0 = wave * 128 + ct * 16;
      v8f o0 = oacc[2 * ct], o1 = oacc[2 * ct + 1];
      for (int k0 = 0; k0 < MB; k0 += 32) {
        v16bf bf  = load_b_gbl(w2e, DIMC, mb * MB + k0, c0, nsub, ko);
        v16bf af0 = load_a_lds(&acts[nsub][k0 + ko]);
        v16bf af1 = load_a_lds(&acts[16 + nsub][k0 + ko]);
        o0 = __builtin_amdgcn_wmma_f32_16x16x32_bf16(false, af0, false, bf,
                                                     (short)0, o0, false, false);
        o1 = __builtin_amdgcn_wmma_f32_16x16x32_bf16(false, af1, false, bf,
                                                     (short)0, o1, false, false);
      }
      oacc[2 * ct] = o0; oacc[2 * ct + 1] = o1;
    }
    __syncthreads();
  }
  // ---- epilogue: y[token] += cw * (out + b2); exactly 2 atomics/element
#pragma unroll
  for (int ct = 0; ct < 8; ++ct) {
    const int c = wave * 128 + ct * 16 + nsub;
    const float bb = b2[e * DIMC + c];
#pragma unroll
    for (int r = 0; r < 2; ++r) {
      const v8f o = oacc[2 * ct + r];
#pragma unroll
      for (int i = 0; i < 8; ++i) {
        int m = i + ko + 16 * r;
        float val = scw[m] * (o[i] + bb);
        atomicAdd(&y[(size_t)stok[m] * DIMC + c], val);
      }
    }
  }
}

// =======================================================================
// Kernel 3: deterministic load-balancing loss reduction.
// =======================================================================
__global__ void lb_kernel(const int* __restrict__ cnt,
                          const float* __restrict__ probs,
                          float* __restrict__ out_lb) {
  __shared__ float se[EXP];
  const int e = threadIdx.x;
  if (e < EXP) {
    float s = 0.f;
    for (int n = 0; n < NTOK; ++n) s += probs[(size_t)n * EXP + e];
    se[e] = ((float)cnt[e] / (float)NTOK) * (s / (float)NTOK);
  }
  __syncthreads();
  if (e == 0) {
    float t = 0.f;
#pragma unroll
    for (int i = 0; i < EXP; ++i) t += se[i];
    *out_lb = 0.01f * (float)EXP * t;
  }
}

// =======================================================================
extern "C" void kernel_launch(void* const* d_in, const int* in_sizes, int n_in,
                              void* d_out, int out_size, void* d_ws, size_t ws_size,
                              hipStream_t stream) {
  const float* x  = (const float*)d_in[0];
  const float* gw = (const float*)d_in[1];
  const float* gb = (const float*)d_in[2];
  const float* w1 = (const float*)d_in[3];
  const float* b1 = (const float*)d_in[4];
  const float* w2 = (const float*)d_in[5];
  const float* b2 = (const float*)d_in[6];
  float* y  = (float*)d_out;                 // [2048,1024] fp32
  float* lb = y + (size_t)NTOK * DIMC;       // scalar, appended

  // workspace layout
  char* ws = (char*)d_ws;
  int*   cnt    = (int*)ws;                                    // 8 ints
  int*   bucket = (int*)(ws + 256);                            // E*N ints
  float* cwb    = (float*)(ws + 256 + (size_t)EXP * NTOK * 4); // E*N floats
  float* probs  = (float*)(ws + 256 + (size_t)EXP * NTOK * 8); // N*E floats

  hipMemsetAsync(d_ws, 0, 256, stream);                               // counters
  hipMemsetAsync(d_out, 0, (size_t)NTOK * DIMC * sizeof(float), stream); // y base

  gate_kernel<<<NTOK, 128, 0, stream>>>(x, gw, gb, cnt, bucket, cwb, probs);
  moe_ffn_kernel<<<EXP * (NTOK / TILE_M), NTHREADS, 0, stream>>>(
      x, w1, b1, w2, b2, cnt, bucket, cwb, y);
  lb_kernel<<<1, 32, 0, stream>>>(cnt, probs, lb);
}
